// MPGNN_LSPE_7980049236115
// MI455X (gfx1250) — matrix operand order repaired and checked by reading, not compiled, not run on hardware
//
#include <hip/hip_runtime.h>
#include <stdint.h>
#include <stddef.h>

#define N_NODES 50000
#define N_EDGES 400000
#define N_GRAPHS 256
#define HID 128
#define N_LAYERS 4

typedef __attribute__((ext_vector_type(16))) _Float16 v16h;
typedef __attribute__((ext_vector_type(8)))  _Float16 v8h;
typedef __attribute__((ext_vector_type(8)))  float    v8f;

// Explicit global address space so loads lower to global_load_b128 (not flat).
#define GAS __attribute__((address_space(1)))
typedef GAS const v8h* gv8h_p;
typedef GAS const v8f* gv8f_p;

struct GemmCfg {
    const _Float16* seg0; const _Float16* seg1; const _Float16* seg2;
    const _Float16* seg3; const _Float16* seg4;
    int mode0, mode1, mode2, mode3, mode4;   // 0 = send[m], 1 = rec[m], 2 = m
};

__device__ __forceinline__ void atomAddF32(float* p, float v) {
    unsafeAtomicAdd(p, v);
}

// A fragment (16x32 f16). Per ISA layout: lanes 0-15 row M=lane, halves kh=0;
// lanes 16-31 row M=lane-16, halves kh=8. VGPR halves [0..7] = K kh..kh+7,
// halves [8..15] = K 16+kh .. 16+kh+7.
__device__ __forceinline__ v16h ld_afrag(const _Float16* row, int kh) {
    v8h lo = *(gv8h_p)(row + kh);
    v8h hi = *(gv8h_p)(row + 16 + kh);
    v16h r;
#pragma unroll
    for (int i = 0; i < 8; ++i) { r[i] = lo[i]; r[8 + i] = hi[i]; }
    return r;
}

// B fragment (32x16 f16). Weights are pre-swizzled so each lane reads two 8-half
// runs: lane holds column n, K = kb..kb+7 and 16+kb..16+kb+7.
__device__ __forceinline__ v16h ld_bfrag(const _Float16* p) {
    v8h lo = *(gv8h_p)(p);
    v8h hi = *(gv8h_p)(p + 16);
    v16h r;
#pragma unroll
    for (int i = 0; i < 8; ++i) { r[i] = lo[i]; r[8 + i] = hi[i]; }
    return r;
}

// Cast + swizzle weights: src fp32 [128][K] (y = x @ w.T), dst f16 in
// B-fragment order: dst[((k>>5)*128 + n)*32 + (k&31)] = w[n][k], zero-padded.
__global__ __launch_bounds__(256) void k_cast_w(const float* __restrict__ w,
                                                _Float16* __restrict__ dst,
                                                int K, int Kpad) {
    int i = blockIdx.x * 256 + threadIdx.x;
    if (i >= Kpad * HID) return;
    int k = i >> 7, n = i & 127;
    float v = (k < K) ? w[(size_t)n * K + k] : 0.0f;
    dst[(((size_t)(k >> 5) * HID + n) << 5) + (k & 31)] = (_Float16)v;
}

// Generic fused gather-GEMM:  out[M,128] = concat(seg_i[idx_i(m)]) @ B + bias
// Block: 256 thr = 8 waves; tile M=64 (4 subtiles), N=128 (wave -> 16 cols).
// NSEG is a template arg so the segment loop fully unrolls and each segment
// base stays a uniform SGPR-pair (saddr-form global loads).
template <int NSEG>
__global__ __launch_bounds__(256) void k_gemm(
    GemmCfg cfg, const int* __restrict__ send, const int* __restrict__ rec,
    const _Float16* __restrict__ Bw, const float* __restrict__ bias,
    int M, int scatter, _Float16* __restrict__ out16, float* __restrict__ agg32) {
    int tid  = threadIdx.x;
    int lane = tid & 31;
    int wave = tid >> 5;
    int nloc = lane & 15;
    int hsel = lane >> 4;
    int kh = hsel * 8;     // A half-group offset
    int kb = hsel * 8;     // B K offset
    int n  = wave * 16 + nloc;
    int m0 = blockIdx.x * 64;

    v8f acc[4];
#pragma unroll
    for (int s = 0; s < 4; ++s)
#pragma unroll
        for (int i = 0; i < 8; ++i) acc[s][i] = 0.0f;

    const _Float16* segp[5] = {cfg.seg0, cfg.seg1, cfg.seg2, cfg.seg3, cfg.seg4};
    const int segm[5] = {cfg.mode0, cfg.mode1, cfg.mode2, cfg.mode3, cfg.mode4};

#pragma unroll
    for (int seg = 0; seg < NSEG; ++seg) {
        const _Float16* sp = segp[seg];
        const int mode = segm[seg];
        // Row offsets for this lane's 4 subtile rows (gather fused here).
        size_t roff[4];
#pragma unroll
        for (int s = 0; s < 4; ++s) {
            int m = m0 + s * 16 + nloc;
            if (m >= M) m = M - 1;                       // clamp; store is guarded
            int idx = (mode == 0) ? send[m] : (mode == 1) ? rec[m] : m;
            roff[s] = (size_t)idx * HID;
        }
        const _Float16* bseg = Bw + (size_t)seg * 4 * HID * 32;  // 4 k-slices / 128
#pragma unroll
        for (int kk = 0; kk < HID; kk += 32) {
            v16h bf = ld_bfrag(bseg + (((size_t)(kk >> 5) * HID + n) << 5) + kb);
#pragma unroll
            for (int s = 0; s < 4; ++s) {
                v16h af = ld_afrag(sp + roff[s] + kk, kh);
                acc[s] = __builtin_amdgcn_wmma_f32_16x16x32_f16(
                    false, af, false, bf, (short)0, acc[s], false, false);
            }
        }
    }

    // Epilogue. C/D layout: lane holds col n, rows M = hsel*8 + r per subtile.
    float bv = bias[n];
#pragma unroll
    for (int s = 0; s < 4; ++s) {
#pragma unroll
        for (int r = 0; r < 8; ++r) {
            int m = m0 + s * 16 + hsel * 8 + r;
            if (m < M) {
                float v = acc[s][r] + bv;
                if (scatter) atomAddF32(&agg32[(size_t)rec[m] * HID + n], v);
                else         out16[(size_t)m * HID + n] = (_Float16)v;
            }
        }
    }
}

// Embedding GEMM: fp32 A [M,KIN] (KIN = 64 or 16, zero-padded to 32*NSLICE).
// KIN/NSLICE are template args so bound guards fold to constants, A loads
// vectorize (2 x b128 fp32 + packed cvt), and the slice loop fully unrolls.
template <int KIN, int NSLICE>
__global__ __launch_bounds__(256) void k_embed(
    const float* __restrict__ A,
    const _Float16* __restrict__ Bw, const float* __restrict__ bias,
    int M, _Float16* __restrict__ out16) {
    int tid  = threadIdx.x;
    int lane = tid & 31;
    int wave = tid >> 5;
    int nloc = lane & 15;
    int hsel = lane >> 4;
    int kh = hsel * 8, kb = hsel * 8;
    int n  = wave * 16 + nloc;
    int m0 = blockIdx.x * 64;

    v8f acc[4];
#pragma unroll
    for (int s = 0; s < 4; ++s)
#pragma unroll
        for (int i = 0; i < 8; ++i) acc[s][i] = 0.0f;

#pragma unroll
    for (int sl = 0; sl < NSLICE; ++sl) {
        v16h bf = ld_bfrag(Bw + (((size_t)sl * HID + n) << 5) + kb);
        const int kkb = sl * 32;
#pragma unroll
        for (int s = 0; s < 4; ++s) {
            int m = m0 + s * 16 + nloc;
            if (m >= M) m = M - 1;
            const float* row = A + (size_t)m * KIN;
            v16h af;
            // halves [0..7]  <- K = kkb + kh + j        (vector load if in range)
            if (kkb + 8 <= KIN) {          // kh+7 <= 15, so kkb+15 < KIN check per half
                v8f lo = *(gv8f_p)(row + kkb + kh);
#pragma unroll
                for (int j = 0; j < 8; ++j) af[j] = (_Float16)lo[j];
            } else {
#pragma unroll
                for (int j = 0; j < 8; ++j) af[j] = (_Float16)0.0f;
            }
            // halves [8..15] <- K = kkb + 16 + kh + j
            if (kkb + 16 + 8 <= KIN) {
                v8f hi = *(gv8f_p)(row + kkb + 16 + kh);
#pragma unroll
                for (int j = 0; j < 8; ++j) af[8 + j] = (_Float16)hi[j];
            } else {
#pragma unroll
                for (int j = 0; j < 8; ++j) af[8 + j] = (_Float16)0.0f;
            }
            acc[s] = __builtin_amdgcn_wmma_f32_16x16x32_f16(
                false, af, false, bf, (short)0, acc[s], false, false);
        }
    }

    float bv = bias[n];
#pragma unroll
    for (int s = 0; s < 4; ++s)
#pragma unroll
        for (int r = 0; r < 8; ++r) {
            int m = m0 + s * 16 + hsel * 8 + r;
            if (m < M) out16[(size_t)m * HID + n] = (_Float16)(acc[s][r] + bv);
        }
}

__global__ __launch_bounds__(256) void k_f2h(const float* __restrict__ in,
                                             _Float16* __restrict__ out, int n) {
    int i = blockIdx.x * 256 + threadIdx.x;
    if (i < n) out[i] = (_Float16)in[i];
}

__global__ __launch_bounds__(256) void k_h2f(const _Float16* __restrict__ in,
                                             float* __restrict__ out, int n) {
    int i = blockIdx.x * 256 + threadIdx.x;
    if (i < n) out[i] = (float)in[i];
}

__global__ __launch_bounds__(256) void k_graph_agg(
    const _Float16* __restrict__ h, const _Float16* __restrict__ p,
    const int* __restrict__ batch, float* __restrict__ hg, float* __restrict__ pg) {
    int i = blockIdx.x * 256 + threadIdx.x;
    if (i >= N_NODES * HID) return;
    int nd = i >> 7, c = i & 127;
    int g = batch[nd];
    atomAddF32(&hg[(size_t)g * HID + c], (float)h[i]);
    atomAddF32(&pg[(size_t)g * HID + c], (float)p[i]);
}

// Readout MLP: one block per graph.
__global__ __launch_bounds__(128) void k_readout(
    const float* __restrict__ hg, const float* __restrict__ pg,
    const float* __restrict__ r1w, const float* __restrict__ r1b,
    const float* __restrict__ r2w, const float* __restrict__ r2b,
    const float* __restrict__ r3w, const float* __restrict__ r3b,
    float* __restrict__ out) {
    __shared__ float hep[2 * HID];
    __shared__ float x1[HID];
    __shared__ float x2[HID / 2];
    int g = blockIdx.x, t = threadIdx.x;
    hep[t]       = hg[(size_t)g * HID + t];
    hep[HID + t] = pg[(size_t)g * HID + t];
    __syncthreads();
    float s = r1b[t];
    for (int i = 0; i < 2 * HID; ++i) s += hep[i] * r1w[(size_t)t * 2 * HID + i];
    x1[t] = fmaxf(s, 0.0f);
    __syncthreads();
    if (t < HID / 2) {
        float s2 = r2b[t];
        for (int i = 0; i < HID; ++i) s2 += x1[i] * r2w[(size_t)t * HID + i];
        x2[t] = fmaxf(s2, 0.0f);
    }
    __syncthreads();
    if (t == 0) {
        float s3 = r3b[0];
        for (int i = 0; i < HID / 2; ++i) s3 += x2[i] * r3w[i];
        out[g] = s3;
    }
}

extern "C" void kernel_launch(void* const* d_in, const int* in_sizes, int n_in,
                              void* d_out, int out_size, void* d_ws, size_t ws_size,
                              hipStream_t stream) {
    const float* h_in = (const float*)d_in[0];
    const float* e_in = (const float*)d_in[1];
    const float* p_in = (const float*)d_in[2];
    const int* edge_index = (const int*)d_in[3];
    const int* send = edge_index;
    const int* rec  = edge_index + N_EDGES;
    const int* batch = (const int*)d_in[4];
    const float* h_embed_w = (const float*)d_in[5];
    const float* h_embed_b = (const float*)d_in[6];
    const float* e_embed_w = (const float*)d_in[7];
    const float* e_embed_b = (const float*)d_in[8];
    const float* p_embed_w = (const float*)d_in[9];
    const float* p_embed_b = (const float*)d_in[10];
    const float* hu_w = (const float*)d_in[11];
    const float* hu_b = (const float*)d_in[12];
    const float* eu_w = (const float*)d_in[13];
    const float* eu_b = (const float*)d_in[14];
    const float* pu_w = (const float*)d_in[15];
    const float* pu_b = (const float*)d_in[16];
    const float* hagg_w = (const float*)d_in[17];
    const float* hagg_b = (const float*)d_in[18];
    const float* pagg_w = (const float*)d_in[19];
    const float* pagg_b = (const float*)d_in[20];
    const float* r1w = (const float*)d_in[21];
    const float* r1b = (const float*)d_in[22];
    const float* r2w = (const float*)d_in[23];
    const float* r2b = (const float*)d_in[24];
    const float* r3w = (const float*)d_in[25];
    const float* r3b = (const float*)d_in[26];
    float* out = (float*)d_out;

    // Bump allocator in workspace (256B aligned).
    size_t off = 0;
    auto alloc = [&](size_t bytes) -> void* {
        void* p = (char*)d_ws + off;
        off = (off + bytes + 255) & ~(size_t)255;
        return p;
    };
    _Float16* h16    = (_Float16*)alloc((size_t)N_NODES * HID * 2);
    _Float16* p16    = (_Float16*)alloc((size_t)N_NODES * HID * 2);
    _Float16* e16    = (_Float16*)alloc((size_t)N_EDGES * HID * 2);
    float*    hagg32 = (float*)   alloc((size_t)N_NODES * HID * 4);
    float*    pagg32 = (float*)   alloc((size_t)N_NODES * HID * 4);
    _Float16* hagg16 = (_Float16*)alloc((size_t)N_NODES * HID * 2);
    _Float16* pagg16 = (_Float16*)alloc((size_t)N_NODES * HID * 2);
    float*    hg     = (float*)   alloc((size_t)N_GRAPHS * HID * 4);
    float*    pg     = (float*)   alloc((size_t)N_GRAPHS * HID * 4);
    const size_t SLICE = (size_t)HID * 32;  // halves per 32-K slice
    _Float16* w_hemb = (_Float16*)alloc(2  * SLICE * 2);
    _Float16* w_eemb = (_Float16*)alloc(1  * SLICE * 2);
    _Float16* w_pemb = (_Float16*)alloc(1  * SLICE * 2);
    _Float16* w_hu   = (_Float16*)alloc((size_t)N_LAYERS * 20 * SLICE * 2);
    _Float16* w_eu   = (_Float16*)alloc((size_t)N_LAYERS * 12 * SLICE * 2);
    _Float16* w_pu   = (_Float16*)alloc((size_t)N_LAYERS * 12 * SLICE * 2);
    _Float16* w_hagg = (_Float16*)alloc((size_t)N_LAYERS * 8  * SLICE * 2);
    _Float16* w_pagg = (_Float16*)alloc((size_t)N_LAYERS * 8  * SLICE * 2);

    dim3 blk(256);
    auto cast_w = [&](const float* src, _Float16* dst, int K, int Kpad) {
        int tot = Kpad * HID;
        k_cast_w<<<dim3((tot + 255) / 256), blk, 0, stream>>>(src, dst, K, Kpad);
    };
    cast_w(h_embed_w, w_hemb, 64, 64);
    cast_w(e_embed_w, w_eemb, 16, 32);
    cast_w(p_embed_w, w_pemb, 16, 32);
    for (int l = 0; l < N_LAYERS; ++l) {
        cast_w(hu_w   + (size_t)l * HID * 640, w_hu   + (size_t)l * 20 * SLICE, 640, 640);
        cast_w(eu_w   + (size_t)l * HID * 384, w_eu   + (size_t)l * 12 * SLICE, 384, 384);
        cast_w(pu_w   + (size_t)l * HID * 384, w_pu   + (size_t)l * 12 * SLICE, 384, 384);
        cast_w(hagg_w + (size_t)l * HID * 256, w_hagg + (size_t)l * 8  * SLICE, 256, 256);
        cast_w(pagg_w + (size_t)l * HID * 256, w_pagg + (size_t)l * 8  * SLICE, 256, 256);
    }

    dim3 gN((N_NODES + 63) / 64), gE((N_EDGES + 63) / 64);
    k_embed<64, 2><<<gN, blk, 0, stream>>>(h_in, w_hemb, h_embed_b, N_NODES, h16);
    k_embed<16, 1><<<gE, blk, 0, stream>>>(e_in, w_eemb, e_embed_b, N_EDGES, e16);
    k_embed<16, 1><<<gN, blk, 0, stream>>>(p_in, w_pemb, p_embed_b, N_NODES, p16);

    const int NH = N_NODES * HID;
    dim3 gNH((NH + 255) / 256);
    for (int l = 0; l < N_LAYERS; ++l) {
        // h_msg = lin(concat(h[s], p[s], h[r], p[r], e)); scatter-sum by rec
        hipMemsetAsync(hagg32, 0, (size_t)NH * 4, stream);
        GemmCfg c1 = {h16, p16, h16, p16, e16, 0, 0, 1, 1, 2};
        k_gemm<5><<<gE, blk, 0, stream>>>(c1, send, rec, w_hu + (size_t)l * 20 * SLICE,
                                          hu_b + (size_t)l * HID, N_EDGES, 1, nullptr, hagg32);
        k_f2h<<<gNH, blk, 0, stream>>>(hagg32, hagg16, NH);
        // h = lin(concat(h, h_msg_agg))   (in place: block reads only own rows)
        GemmCfg c2 = {h16, hagg16, nullptr, nullptr, nullptr, 2, 2, 2, 2, 2};
        k_gemm<2><<<gN, blk, 0, stream>>>(c2, send, rec, w_hagg + (size_t)l * 8 * SLICE,
                                          hagg_b + (size_t)l * HID, N_NODES, 0, h16, nullptr);
        // e = lin(concat(h[s], h[r], e))  (in place)
        GemmCfg c3 = {h16, h16, e16, nullptr, nullptr, 0, 1, 2, 2, 2};
        k_gemm<3><<<gE, blk, 0, stream>>>(c3, send, rec, w_eu + (size_t)l * 12 * SLICE,
                                          eu_b + (size_t)l * HID, N_EDGES, 0, e16, nullptr);
        // p_msg = lin(concat(p[s], p[r], e_new)); scatter-sum by rec
        hipMemsetAsync(pagg32, 0, (size_t)NH * 4, stream);
        GemmCfg c4 = {p16, p16, e16, nullptr, nullptr, 0, 1, 2, 2, 2};
        k_gemm<3><<<gE, blk, 0, stream>>>(c4, send, rec, w_pu + (size_t)l * 12 * SLICE,
                                          pu_b + (size_t)l * HID, N_EDGES, 1, nullptr, pagg32);
        k_f2h<<<gNH, blk, 0, stream>>>(pagg32, pagg16, NH);
        // p = lin(concat(p, p_msg_agg))   (in place)
        GemmCfg c5 = {p16, pagg16, nullptr, nullptr, nullptr, 2, 2, 2, 2, 2};
        k_gemm<2><<<gN, blk, 0, stream>>>(c5, send, rec, w_pagg + (size_t)l * 8 * SLICE,
                                          pagg_b + (size_t)l * HID, N_NODES, 0, p16, nullptr);
    }

    // Readout
    hipMemsetAsync(hg, 0, (size_t)N_GRAPHS * HID * 4, stream);
    hipMemsetAsync(pg, 0, (size_t)N_GRAPHS * HID * 4, stream);
    k_graph_agg<<<gNH, blk, 0, stream>>>(h16, p16, batch, hg, pg);
    k_readout<<<dim3(N_GRAPHS), dim3(128), 0, stream>>>(hg, pg, r1w, r1b, r2w, r2b,
                                                        r3w, r3b, out);
    // Second output: final p (fp32), concatenated after the 256 graph outputs
    k_h2f<<<gNH, blk, 0, stream>>>(p16, out + N_GRAPHS, NH);
}
